// LlamaAttention_81466939671155
// MI455X (gfx1250) — matrix operand order repaired and checked by reading, not compiled, not run on hardware
//
#include <hip/hip_runtime.h>
#include <hip/hip_bf16.h>
#include <math.h>

// ---------------- problem constants (match reference) ----------------
constexpr int kB    = 4;
constexpr int kQ    = 512;
constexpr int kHIST = 1536;
constexpr int kKV   = kHIST + kQ;   // 2048
constexpr int kHID  = 4096;
constexpr int kNH   = 32;
constexpr int kHD   = 128;
constexpr int kBS   = 64;
constexpr int kMAXB = kKV / kBS;    // 32
constexpr int kT    = kB * kQ;      // 2048

typedef __attribute__((ext_vector_type(16))) __bf16 v16bf;
typedef __attribute__((ext_vector_type(8)))  float  v8f;
typedef __attribute__((ext_vector_type(4)))  int    v4i;

// ------- gfx1250 async global->LDS path (guarded; fallback = VGPR path) ----
#if defined(__AMDGCN__) && __has_builtin(__builtin_amdgcn_global_load_async_to_lds_b128)
#define USE_ASYNC_LDS 1
typedef __attribute__((address_space(1))) v4i gv4i;   // global int4
typedef __attribute__((address_space(3))) v4i lv4i;   // LDS int4
#else
#define USE_ASYNC_LDS 0
#endif

static __device__ __forceinline__ void wait_async_zero() {
#if defined(__AMDGCN__)
#if __has_builtin(__builtin_amdgcn_s_wait_asynccnt)
  __builtin_amdgcn_s_wait_asynccnt(0);
#else
  asm volatile("s_wait_asynccnt 0" ::: "memory");
#endif
#endif
}

// ---------------------------------------------------------------------
// 1) f32 -> bf16 copy (grid-stride)
// ---------------------------------------------------------------------
__global__ void cvt_bf16_kernel(const float* __restrict__ src,
                                __bf16* __restrict__ dst, long n) {
  long i = (long)blockIdx.x * blockDim.x + threadIdx.x;
  long stride = (long)gridDim.x * blockDim.x;
  for (; i < n; i += stride) dst[i] = (__bf16)src[i];
}

// ---------------------------------------------------------------------
// 2) GEMM: C[M,N](f32) = A[M,K](bf16) @ W[N,K](f32)^T   via WMMA bf16
//    128x128 tile / block, 8 waves (wave = 64x32 subtile = 4x2 wmma accs)
//    A tile: async global->LDS (ASYNCcnt); W tile: load+cvt via VGPRs
// ---------------------------------------------------------------------
#define TM 128
#define TN 128
#define TK 32
#define LPAD 8   // row stride = 40 bf16 = 80 B (multiple of 16 B for B128 LDS)

__global__ __launch_bounds__(256) void gemm_bf16w_kernel(
    const __bf16* __restrict__ A, const float* __restrict__ W,
    float* __restrict__ C, int M, int N, int K) {
  __shared__ __bf16 As[TM][TK + LPAD];
  __shared__ __bf16 Bs[TN][TK + LPAD];

  const int tid  = threadIdx.x;
  const int lane = tid & 31;
  const int wave = tid >> 5;
  const int half = lane >> 4;
  const int mrow = lane & 15;
  const int bm = blockIdx.y * TM;
  const int bn = blockIdx.x * TN;
  const int wm = (wave & 1) * 64;   // 4 m-subtiles of 16
  const int wn = (wave >> 1) * 32;  // 2 n-subtiles of 16

  v8f acc[4][2];
#pragma unroll
  for (int mt = 0; mt < 4; ++mt)
#pragma unroll
    for (int nt = 0; nt < 2; ++nt) {
      v8f z = {0.f,0.f,0.f,0.f,0.f,0.f,0.f,0.f};
      acc[mt][nt] = z;
    }

  const int ldrow = tid >> 1;          // 0..127
  const int ldcol = (tid & 1) * 16;    // 0 or 16

  for (int k0 = 0; k0 < K; k0 += TK) {
    const __bf16* ap = A + (size_t)(bm + ldrow) * K + k0 + ldcol;
    const float*  wp = W + (size_t)(bn + ldrow) * K + k0 + ldcol;

#if USE_ASYNC_LDS
    // fire-and-forget: LDS[As] <= global A tile, 2 x 16 B per thread
    __builtin_amdgcn_global_load_async_to_lds_b128(
        (gv4i*)ap, (lv4i*)&As[ldrow][ldcol], 0, 0);
    __builtin_amdgcn_global_load_async_to_lds_b128(
        (gv4i*)(ap + 8), (lv4i*)&As[ldrow][ldcol + 8], 0, 0);
#else
#pragma unroll
    for (int i = 0; i < 16; ++i) As[ldrow][ldcol + i] = ap[i];
#endif
    // W tile needs f32->bf16 conversion: overlap with the async copy above
#pragma unroll
    for (int i = 0; i < 16; ++i) Bs[ldrow][ldcol + i] = (__bf16)wp[i];

    if (k0 + TK < K) {                 // prefetch next K-tile (global_prefetch_b8)
      __builtin_prefetch(ap + TK, 0, 3);
      __builtin_prefetch(wp + TK, 0, 3);
    }
#if USE_ASYNC_LDS
    wait_async_zero();                 // drain ASYNCcnt before the barrier
#endif
    __syncthreads();

#pragma unroll
    for (int nt = 0; nt < 2; ++nt) {
      v16bf bfrag;                      // B 32x16: lane -> col N=mrow, K contiguous per half
      const __bf16* bp = &Bs[wn + nt * 16 + mrow][half * 16];
#pragma unroll
      for (int i = 0; i < 16; ++i) bfrag[i] = bp[i];
#pragma unroll
      for (int mt = 0; mt < 4; ++mt) {
        v16bf afrag;                    // A 16x32: lane -> row M=mrow, K contiguous per half
        const __bf16* apl = &As[wm + mt * 16 + mrow][0];
#pragma unroll
        for (int i = 0; i < 8; ++i) {
          afrag[i]     = apl[half * 16 + i];
          afrag[8 + i] = apl[half * 16 + 8 + i];
        }
        acc[mt][nt] = __builtin_amdgcn_wmma_f32_16x16x32_bf16(
            false, afrag, false, bfrag, (short)0, acc[mt][nt], false, false);
      }
    }
    __syncthreads();
  }

  // D layout: element (vgpr r, lane) -> M = r + 8*half, N = mrow
#pragma unroll
  for (int mt = 0; mt < 4; ++mt)
#pragma unroll
    for (int nt = 0; nt < 2; ++nt)
#pragma unroll
      for (int r = 0; r < 8; ++r) {
        int row = bm + wm + mt * 16 + r + 8 * half;
        int col = bn + wn + nt * 16 + mrow;
        C[(size_t)row * N + col] = acc[mt][nt][r];
      }
}

// ---------------------------------------------------------------------
// 3) RoPE + layout kernels
// ---------------------------------------------------------------------
__global__ void rope_q_kernel(const float* __restrict__ qt,
                              const int* __restrict__ pos_ids,
                              __bf16* __restrict__ qb) {
  long id = (long)blockIdx.x * blockDim.x + threadIdx.x;
  if (id >= (long)kT * kNH * 64) return;
  int d = id & 63; int h = (id >> 6) & 31; int t = (int)(id >> 11);
  int b = t / kQ, qi = t % kQ;
  int pos = pos_ids[b * kQ + qi];
  float inv = __powf(10000.0f, -(float)(2 * d) / (float)kHD);
  float s, c; __sincosf((float)pos * inv, &s, &c);
  const float* src = qt + (size_t)t * kHID + h * kHD;
  float x0 = src[d], x1 = src[d + 64];
  __bf16* dst = qb + ((size_t)(b * kNH + h) * kQ + qi) * kHD;
  dst[d]      = (__bf16)(x0 * c - x1 * s);
  dst[d + 64] = (__bf16)(x1 * c + x0 * s);
}

__global__ void rope_k_kernel(const float* __restrict__ kt,
                              const int* __restrict__ pos_ids,
                              __bf16* __restrict__ kseq) {
  long id = (long)blockIdx.x * blockDim.x + threadIdx.x;
  if (id >= (long)kT * kNH * 64) return;
  int d = id & 63; int h = (id >> 6) & 31; int t = (int)(id >> 11);
  int b = t / kQ, qi = t % kQ;
  int pos = pos_ids[b * kQ + qi];
  float inv = __powf(10000.0f, -(float)(2 * d) / (float)kHD);
  float s, c; __sincosf((float)pos * inv, &s, &c);
  const float* src = kt + (size_t)t * kHID + h * kHD;
  float x0 = src[d], x1 = src[d + 64];
  __bf16* dst = kseq + ((size_t)(b * kNH + h) * kKV + pos) * kHD;
  dst[d]      = (__bf16)(x0 * c - x1 * s);
  dst[d + 64] = (__bf16)(x1 * c + x0 * s);
}

__global__ void scatter_v_kernel(const float* __restrict__ vt,
                                 const int* __restrict__ pos_ids,
                                 __bf16* __restrict__ vT) {
  long id = (long)blockIdx.x * blockDim.x + threadIdx.x;
  if (id >= (long)kT * kHID) return;
  int d = id & 127; int h = (id >> 7) & 31; int t = (int)(id >> 12);
  int b = t / kQ, qi = t % kQ;
  int pos = pos_ids[b * kQ + qi];
  vT[((size_t)(b * kNH + h) * kHD + d) * kKV + pos] =
      (__bf16)vt[(size_t)t * kHID + h * kHD + d];
}

// gather history (kv < HIST) from paged caches via block table
__global__ void gather_k_kernel(const float* __restrict__ kc,
                                const int* __restrict__ bofs,
                                __bf16* __restrict__ kseq) {
  long id = (long)blockIdx.x * blockDim.x + threadIdx.x;
  if (id >= (long)kB * kHIST * kNH * kHD) return;
  int d = id & 127; int h = (id >> 7) & 31;
  long rest = id >> 12;
  int kv = (int)(rest % kHIST); int b = (int)(rest / kHIST);
  int blk = bofs[b * kMAXB + (kv >> 6)];
  int slot = kv & 63;
  float v = kc[(((size_t)blk * kBS + slot) * kNH + h) * kHD + d];
  kseq[((size_t)(b * kNH + h) * kKV + kv) * kHD + d] = (__bf16)v;
}

__global__ void gather_v_kernel(const float* __restrict__ vc,
                                const int* __restrict__ bofs,
                                __bf16* __restrict__ vT) {
  long id = (long)blockIdx.x * blockDim.x + threadIdx.x;
  if (id >= (long)kB * kHIST * kNH * kHD) return;
  int d = id & 127; int h = (id >> 7) & 31;
  long rest = id >> 12;
  int kv = (int)(rest % kHIST); int b = (int)(rest / kHIST);
  int blk = bofs[b * kMAXB + (kv >> 6)];
  int slot = kv & 63;
  float v = vc[(((size_t)blk * kBS + slot) * kNH + h) * kHD + d];
  vT[((size_t)(b * kNH + h) * kHD + d) * kKV + kv] = (__bf16)v;
}

// ---------------------------------------------------------------------
// 4) Fused flash attention: 1 wave per (b, h, 16-row q tile)
// ---------------------------------------------------------------------
__global__ __launch_bounds__(32) void attn_kernel(
    const __bf16* __restrict__ Qb,    // [B,NH,Q,HD]
    const __bf16* __restrict__ Kseq,  // [B,NH,KV,HD]
    const __bf16* __restrict__ Vt,    // [B,NH,HD,KV]
    __bf16* __restrict__ Ao) {        // [T,HID] == [B,Q,NH,HD]
  __shared__ __bf16 Pl[16][TK + LPAD];

  const int lane = threadIdx.x;
  const int half = lane >> 4;
  const int mrow = lane & 15;
  const int qt = blockIdx.x % (kQ / 16);
  const int hb = blockIdx.x / (kQ / 16);
  const int h = hb % kNH;
  const int b = hb / kNH;
  const int q0 = qt * 16;
  const int qpos = kHIST + q0;

  const __bf16* qrow  = Qb   + ((size_t)(b * kNH + h) * kQ + q0 + mrow) * kHD;
  const __bf16* kbase = Kseq + ((size_t)(b * kNH + h) * kKV) * kHD;
  const __bf16* vbase = Vt   + ((size_t)(b * kNH + h) * kHD) * kKV;

  v16bf aQ[4];
#pragma unroll
  for (int f = 0; f < 4; ++f)
#pragma unroll
    for (int i = 0; i < 8; ++i) {
      aQ[f][i]     = qrow[f * 32 + half * 16 + i];
      aQ[f][8 + i] = qrow[f * 32 + half * 16 + 8 + i];
    }

  v8f o[8];
#pragma unroll
  for (int n = 0; n < 8; ++n) {
    v8f z = {0.f,0.f,0.f,0.f,0.f,0.f,0.f,0.f};
    o[n] = z;
  }
  float mi[8], li[8];
#pragma unroll
  for (int r = 0; r < 8; ++r) { mi[r] = -1e30f; li[r] = 0.f; }

  const float sc = 0.08838834764831845f;  // 1/sqrt(128)
  const int kv_end = qpos + 16;           // causal bound for this tile

  for (int kv0 = 0; kv0 < kv_end; kv0 += 32) {
    v8f s0 = {0.f,0.f,0.f,0.f,0.f,0.f,0.f,0.f};
    v8f s1 = {0.f,0.f,0.f,0.f,0.f,0.f,0.f,0.f};
#pragma unroll
    for (int f = 0; f < 4; ++f) {
      v16bf b0, b1;
      const __bf16* kp0 = kbase + (size_t)(kv0 + mrow) * kHD + f * 32 + half * 16;
      const __bf16* kp1 = kbase + (size_t)(kv0 + 16 + mrow) * kHD + f * 32 + half * 16;
#pragma unroll
      for (int i = 0; i < 16; ++i) { b0[i] = kp0[i]; b1[i] = kp1[i]; }
      s0 = __builtin_amdgcn_wmma_f32_16x16x32_bf16(false, aQ[f], false, b0,
                                                   (short)0, s0, false, false);
      s1 = __builtin_amdgcn_wmma_f32_16x16x32_bf16(false, aQ[f], false, b1,
                                                   (short)0, s1, false, false);
    }

    float alpha[8];
#pragma unroll
    for (int r = 0; r < 8; ++r) {
      int M = r + 8 * half;
      float v0 = s0[r] * sc, v1 = s1[r] * sc;
      if (kv0 + mrow > qpos + M)      v0 = -1e30f;   // causal mask
      if (kv0 + 16 + mrow > qpos + M) v1 = -1e30f;
      float t = fmaxf(v0, v1);                        // row max over 16 lanes
      t = fmaxf(t, __shfl_xor(t, 1, 32));
      t = fmaxf(t, __shfl_xor(t, 2, 32));
      t = fmaxf(t, __shfl_xor(t, 4, 32));
      t = fmaxf(t, __shfl_xor(t, 8, 32));
      float nm = fmaxf(mi[r], t);
      alpha[r] = __expf(mi[r] - nm);
      mi[r] = nm;
      float p0 = __expf(v0 - nm), p1 = __expf(v1 - nm);
      float rs = p0 + p1;
      rs += __shfl_xor(rs, 1, 32);
      rs += __shfl_xor(rs, 2, 32);
      rs += __shfl_xor(rs, 4, 32);
      rs += __shfl_xor(rs, 8, 32);
      li[r] = li[r] * alpha[r] + rs;
      Pl[M][mrow]      = (__bf16)p0;  // D-layout -> LDS (re-layout to A-frag)
      Pl[M][16 + mrow] = (__bf16)p1;
    }
#pragma unroll
    for (int n = 0; n < 8; ++n)
#pragma unroll
      for (int r = 0; r < 8; ++r) o[n][r] *= alpha[r];
    __syncthreads();

    v16bf aP;
    const __bf16* pp = &Pl[mrow][0];
#pragma unroll
    for (int i = 0; i < 8; ++i) {
      aP[i]     = pp[half * 16 + i];
      aP[8 + i] = pp[half * 16 + 8 + i];
    }
#pragma unroll
    for (int n = 0; n < 8; ++n) {   // P[16x32] @ V[32x16] per 16-dim slice
      v16bf bv;
      const __bf16* vp = vbase + (size_t)(n * 16 + mrow) * kKV + kv0 + half * 16;
#pragma unroll
      for (int i = 0; i < 16; ++i) bv[i] = vp[i];
      o[n] = __builtin_amdgcn_wmma_f32_16x16x32_bf16(false, aP, false, bv,
                                                     (short)0, o[n], false, false);
    }
    __syncthreads();
  }

#pragma unroll
  for (int r = 0; r < 8; ++r) {
    int M = r + 8 * half;
    float inv = 1.0f / li[r];
    size_t t = (size_t)(b * kQ + q0 + M);
#pragma unroll
    for (int n = 0; n < 8; ++n)
      Ao[t * kHID + h * kHD + n * 16 + mrow] = (__bf16)(o[n][r] * inv);
  }
}

// ---------------------------------------------------------------------
// launch
// ---------------------------------------------------------------------
extern "C" void kernel_launch(void* const* d_in, const int* in_sizes, int n_in,
                              void* d_out, int out_size, void* d_ws, size_t ws_size,
                              hipStream_t stream) {
  const float* x    = (const float*)d_in[0];
  const float* wq   = (const float*)d_in[1];
  const float* wk   = (const float*)d_in[2];
  const float* wv   = (const float*)d_in[3];
  const float* wo   = (const float*)d_in[4];
  const float* kc   = (const float*)d_in[5];
  const float* vc   = (const float*)d_in[6];
  const int*   bofs = (const int*)d_in[7];
  const int*   pids = (const int*)d_in[8];
  float* out = (float*)d_out;

  char* ws = (char*)d_ws;
  size_t off = 0;
  __bf16* xb   = (__bf16*)(ws + off); off += (size_t)kT * kHID * 2;
  float* qtmp  = (float*)(ws + off);  off += (size_t)kT * kHID * 4;
  float* ktmp  = (float*)(ws + off);  off += (size_t)kT * kHID * 4;
  float* vtmp  = (float*)(ws + off);  off += (size_t)kT * kHID * 4;
  __bf16* qb   = (__bf16*)(ws + off); off += (size_t)kB * kNH * kQ * kHD * 2;
  __bf16* kseq = (__bf16*)(ws + off); off += (size_t)kB * kNH * kKV * kHD * 2;
  __bf16* vT   = (__bf16*)(ws + off); off += (size_t)kB * kNH * kHD * kKV * 2;
  __bf16* ao   = (__bf16*)(ws + off); off += (size_t)kT * kHID * 2;

  // 1) x -> bf16
  {
    long n = (long)kT * kHID;
    cvt_bf16_kernel<<<dim3((unsigned)((n + 255) / 256)), dim3(256), 0, stream>>>(x, xb, n);
  }
  // 2) QKV projections (WMMA bf16)
  dim3 gg(kHID / TN, kT / TM), gb(256);
  gemm_bf16w_kernel<<<gg, gb, 0, stream>>>(xb, wq, qtmp, kT, kHID, kHID);
  gemm_bf16w_kernel<<<gg, gb, 0, stream>>>(xb, wk, ktmp, kT, kHID, kHID);
  gemm_bf16w_kernel<<<gg, gb, 0, stream>>>(xb, wv, vtmp, kT, kHID, kHID);
  // 3) RoPE + layout
  {
    long n = (long)kT * kNH * 64;
    unsigned blocks = (unsigned)((n + 255) / 256);
    rope_q_kernel<<<blocks, 256, 0, stream>>>(qtmp, pids, qb);
    rope_k_kernel<<<blocks, 256, 0, stream>>>(ktmp, pids, kseq);
  }
  {
    long n = (long)kT * kHID;
    scatter_v_kernel<<<(unsigned)((n + 255) / 256), 256, 0, stream>>>(vtmp, pids, vT);
  }
  {
    long n = (long)kB * kHIST * kNH * kHD;
    unsigned blocks = (unsigned)((n + 255) / 256);
    gather_k_kernel<<<blocks, 256, 0, stream>>>(kc, bofs, kseq);
    gather_v_kernel<<<blocks, 256, 0, stream>>>(vc, bofs, vT);
  }
  // 4) fused flash attention
  attn_kernel<<<dim3(kB * kNH * (kQ / 16)), dim3(32), 0, stream>>>(qb, kseq, vT, ao);
  // 5) output projection -> d_out (f32)
  gemm_bf16w_kernel<<<gg, gb, 0, stream>>>(ao, wo, out, kT, kHID, kHID);
}
